// MultiLatentAttention_28656021799424
// MI455X (gfx1250) — compile-verified, matrix operand order checked by reading, not run on hardware
//
#include <hip/hip_runtime.h>
#include <cstdint>
#include <cstddef>

// ---------------- problem constants ----------------
#define B_    2
#define T_    1024
#define D_    2048
#define H_    16
#define DN_   128
#define DR_   64
#define DV_   128
#define QR_   1536
#define KVR_  512
#define BT_   (B_ * T_)

typedef __attribute__((ext_vector_type(16))) __bf16 bf16x16;
typedef __attribute__((ext_vector_type(8)))  __bf16 bf16x8;
typedef __attribute__((ext_vector_type(8)))  float  f32x8;

typedef __attribute__((ext_vector_type(4))) unsigned int tdm_u32x4;
typedef __attribute__((ext_vector_type(8))) int          tdm_i32x8;
typedef __attribute__((ext_vector_type(4))) int          tdm_i32x4;

#if defined(__has_builtin)
#if __has_builtin(__builtin_amdgcn_tensor_load_to_lds) && \
    __has_builtin(__builtin_amdgcn_s_wait_tensorcnt)
#define USE_TDM 1
#endif
#endif
#ifndef USE_TDM
#define USE_TDM 0
#endif

// ---------------- WMMA helpers ----------------
__device__ __forceinline__ f32x8 wmma_bf16(bf16x16 a, bf16x16 b, f32x8 c) {
  // 8 args: (neg_a, A, neg_b, B, c_mod, C, reuse_a, reuse_b)
  return __builtin_amdgcn_wmma_f32_16x16x32_bf16(false, a, false, b, (short)0, c,
                                                 false, false);
}

// Load one 16x32 bf16 A/B fragment row for this lane.
// Lane l holds matrix row (l%16); halves j<8 are K = (l/16)*8 + j,
// halves j>=8 are K = 16 + (l/16)*8 + (j-8). Both are contiguous 16B runs.
__device__ __forceinline__ bf16x16 load_frag(const __bf16* p, int lh) {
  union { bf16x16 v; bf16x8 h[2]; } u;
  const __bf16* q = p + lh * 8;
  u.h[0] = *(const bf16x8*)(q);
  u.h[1] = *(const bf16x8*)(q + 16);
  return u.v;
}

#if USE_TDM
// Tensor Data Mover: DMA a [rows x 64] bf16 tile (row stride K elements) from
// global memory into LDS at lds_off, inserting 16B padding every 128B so the
// LDS row stride is 144B (= 72 halves) -- identical to As/Bs layout below.
// D# layout per CDNA5 ISA.
__device__ __forceinline__ void tdm_load_2d(const __bf16* gsrc, unsigned int lds_off,
                                            int rows, int K) {
  const unsigned long long ga = (unsigned long long)(uintptr_t)gsrc;
  tdm_u32x4 g0;
  g0[0] = 1u;                                           // count=1 (valid descriptor)
  g0[1] = lds_off;                                      // lds_addr [63:32]
  g0[2] = (unsigned int)ga;                             // global_addr [95:64]
  g0[3] = (unsigned int)((ga >> 32) & 0x01ffffffu) | (2u << 30);  // addr hi + type=2
  tdm_i32x8 g1;
  // data_size=1 (2B), pad_enable, pad_interval=4 (32 DWORDs), pad_amount=3 (4 DWORDs)
  g1[0] = (int)((1u << 16) | (1u << 20) | (4u << 22) | (3u << 25));
  g1[1] = (int)(((unsigned int)K & 0xffffu) << 16);     // tensor_dim0 lo16 @ [63:48]
  g1[2] = (int)((((unsigned int)K >> 16) & 0xffffu) |   // tensor_dim0 hi16
                (((unsigned int)rows & 0xffffu) << 16));// tensor_dim1 lo16
  g1[3] = (int)((((unsigned int)rows >> 16) & 0xffffu) |
                (64u << 16));                           // tile_dim0 = 64 elements
  g1[4] = (int)((unsigned int)rows & 0xffffu);          // tile_dim1 = rows, tile_dim2=0
  g1[5] = (int)(unsigned int)K;                         // tensor_dim0_stride lo32
  g1[6] = 0;                                            // stride hi16, dim1_stride lo16
  g1[7] = 0;
  const tdm_i32x4 z4 = {0, 0, 0, 0};
#if __clang_major__ >= 23
  const tdm_i32x8 z8 = {0, 0, 0, 0, 0, 0, 0, 0};
  __builtin_amdgcn_tensor_load_to_lds(g0, g1, z4, z4, z8, 0);
#else
  __builtin_amdgcn_tensor_load_to_lds(g0, g1, z4, z4, 0);
#endif
}
#endif  // USE_TDM

// ---------------- elementwise converts ----------------
__global__ void f32_to_bf16_kernel(const float* __restrict__ in,
                                   __bf16* __restrict__ out, long n) {
  long i = (long)blockIdx.x * blockDim.x + threadIdx.x;
  if (i < n) out[i] = (__bf16)in[i];
}

// W[K,N] f32 -> Wt[N,K] bf16 (tiled transpose through LDS)
__global__ __launch_bounds__(256) void wtrans_kernel(const float* __restrict__ W,
                                                     __bf16* __restrict__ Wt,
                                                     int K, int N) {
  __shared__ float tile[32][33];
  const int tx = threadIdx.x & 31, ty = threadIdx.x >> 5;
  const int n0 = blockIdx.x * 32, k0 = blockIdx.y * 32;
#pragma unroll
  for (int i = 0; i < 4; ++i)
    tile[ty + i * 8][tx] = W[(size_t)(k0 + ty + i * 8) * N + n0 + tx];
  __syncthreads();
#pragma unroll
  for (int i = 0; i < 4; ++i)
    Wt[(size_t)(n0 + ty + i * 8) * K + k0 + tx] = (__bf16)tile[tx][ty + i * 8];
}

// ---------------- bf16 WMMA GEMM: C[M,N] f32 = A[M,K] * Bt[N,K] ----------------
// 256 threads = 8 waves in 4(M) x 2(N); block tile 128(M) x 64(N);
// wave tile 32x32 (4 accumulators); K staged 64 deep in LDS.
// Slab staging via the Tensor Data Mover (one descriptor per slab, issued by
// wave 0, TDM pad feature reproduces the 144B LDS row stride); fragment loads
// are ds_load_b128; 8 WMMAs per stage per wave.
__global__ __launch_bounds__(256) void gemm_bf16_kernel(
    const __bf16* __restrict__ A, const __bf16* __restrict__ Bt,
    float* __restrict__ C, int M, int N, int K) {
  __shared__ __align__(16) __bf16 As[128][72];  // 144B row stride (16B aligned)
  __shared__ __align__(16) __bf16 Bs[64][72];
  const int tid = threadIdx.x;
  const int lane = tid & 31, wv = tid >> 5;
  const int wm = wv >> 1, wn = wv & 1;
  const int l16 = lane & 15, lh = lane >> 4;
  const int m0 = blockIdx.y * 128, n0 = blockIdx.x * 64;

  f32x8 acc00 = {}, acc01 = {}, acc10 = {}, acc11 = {};

  for (int kb = 0; kb < K; kb += 64) {
#if USE_TDM
    if (wv == 0) {
      tdm_load_2d(A + (size_t)m0 * K + kb, (unsigned int)(uintptr_t)&As[0][0], 128, K);
      tdm_load_2d(Bt + (size_t)n0 * K + kb, (unsigned int)(uintptr_t)&Bs[0][0], 64, K);
      __builtin_amdgcn_s_wait_tensorcnt(0);
    }
    if (kb + 64 < K) {  // pull next slab toward L2 while DMA/compute runs
      __builtin_prefetch(A + (size_t)(m0 + (tid >> 3)) * K + kb + 64, 0, 3);
      __builtin_prefetch(Bt + (size_t)(n0 + (tid >> 3 & 63)) * K + kb + 64, 0, 3);
    }
#else
    // fallback: cooperative vector-load staging (6 x 16B per thread)
#pragma unroll
    for (int i = 0; i < 4; ++i) {
      const int idx = tid + i * 256;
      const int row = idx >> 3, col = (idx & 7) * 8;
      *(bf16x8*)&As[row][col] =
          *(const bf16x8*)(A + (size_t)(m0 + row) * K + kb + col);
    }
#pragma unroll
    for (int i = 0; i < 2; ++i) {
      const int idx = tid + i * 256;
      const int row = idx >> 3, col = (idx & 7) * 8;
      *(bf16x8*)&Bs[row][col] =
          *(const bf16x8*)(Bt + (size_t)(n0 + row) * K + kb + col);
    }
#endif
    __syncthreads();
#pragma unroll
    for (int kc = 0; kc < 2; ++kc) {
      bf16x16 a0 = load_frag(&As[wm * 32 + l16][kc * 32], lh);
      bf16x16 a1 = load_frag(&As[wm * 32 + 16 + l16][kc * 32], lh);
      bf16x16 b0 = load_frag(&Bs[wn * 32 + l16][kc * 32], lh);
      bf16x16 b1 = load_frag(&Bs[wn * 32 + 16 + l16][kc * 32], lh);
      acc00 = wmma_bf16(a0, b0, acc00);
      acc01 = wmma_bf16(a0, b1, acc01);
      acc10 = wmma_bf16(a1, b0, acc10);
      acc11 = wmma_bf16(a1, b1, acc11);
    }
    __syncthreads();
  }
  // C/D layout: VGPR r holds row r (lanes 0-15) / row 8+r (lanes 16-31), col = lane%16
  const int ncol = n0 + wn * 32 + l16;
  const int mr = m0 + wm * 32 + 8 * lh;
#pragma unroll
  for (int r = 0; r < 8; ++r) C[(size_t)(mr + r) * N + ncol] = acc00[r];
#pragma unroll
  for (int r = 0; r < 8; ++r) C[(size_t)(mr + r) * N + ncol + 16] = acc01[r];
#pragma unroll
  for (int r = 0; r < 8; ++r) C[(size_t)(mr + 16 + r) * N + ncol] = acc10[r];
#pragma unroll
  for (int r = 0; r < 8; ++r) C[(size_t)(mr + 16 + r) * N + ncol + 16] = acc11[r];
}

// ---------------- RoPE ----------------
// q_r: f32 [B,T,H,DR] -> bf16 same layout, rotate-half, base 10000
__global__ void rope_q_kernel(const float* __restrict__ in,
                              const int* __restrict__ pos,
                              __bf16* __restrict__ out) {
  const int idx = blockIdx.x * blockDim.x + threadIdx.x;  // B*T*H*(DR/2) = 1M
  const int i = idx & 31;
  const int h = (idx >> 5) & (H_ - 1);
  const int t = (idx >> 9) & (T_ - 1);
  const int b = idx >> 19;
  if (b >= B_) return;
  const size_t base = ((size_t)(b * T_ + t) * H_ + h) * DR_;
  const float ang = (float)pos[t] * __powf(10000.0f, -(2.0f * i) / (float)DR_);
  float s, c;
  __sincosf(ang, &s, &c);
  const float x1 = in[base + i], x2 = in[base + i + DR_ / 2];
  out[base + i] = (__bf16)(x1 * c - x2 * s);
  out[base + i + DR_ / 2] = (__bf16)(x1 * s + x2 * c);
}

// k_r: f32 [B,T,DR] -> bf16 same layout
__global__ void rope_k_kernel(const float* __restrict__ in,
                              const int* __restrict__ pos,
                              __bf16* __restrict__ out) {
  const int idx = blockIdx.x * blockDim.x + threadIdx.x;  // B*T*(DR/2) = 64K
  const int i = idx & 31;
  const int t = (idx >> 5) & (T_ - 1);
  const int b = idx >> 15;
  if (b >= B_) return;
  const size_t base = (size_t)(b * T_ + t) * DR_;
  const float ang = (float)pos[t] * __powf(10000.0f, -(2.0f * i) / (float)DR_);
  float s, c;
  __sincosf(ang, &s, &c);
  const float x1 = in[base + i], x2 = in[base + i + DR_ / 2];
  out[base + i] = (__bf16)(x1 * c - x2 * s);
  out[base + i + DR_ / 2] = (__bf16)(x1 * s + x2 * c);
}

// v_c f32 [B,T,H,DV] -> Vt bf16 [B,H,DV,T] (so V columns become contiguous K for WMMA-B)
__global__ void vtrans_kernel(const float* __restrict__ vc,
                              __bf16* __restrict__ vt) {
  const int idx = blockIdx.x * blockDim.x + threadIdx.x;  // B*H*DV*T = 4M
  const int t = idx & (T_ - 1);
  const int d = (idx >> 10) & (DV_ - 1);
  const int h = (idx >> 17) & (H_ - 1);
  const int b = idx >> 21;
  if (b >= B_) return;
  vt[idx] = (__bf16)vc[((size_t)(b * T_ + t) * H_ + h) * DV_ + d];
}

// ---------------- flash attention (causal, online softmax) ----------------
// grid (T/64, H, B); 128 threads = 4 waves; wave owns 16 query rows.
// Per 32-key step: 12 QK^T WMMAs + 8 P*V WMMAs.
__global__ __launch_bounds__(128) void mla_attn_kernel(
    const __bf16* __restrict__ qc,   // [B,T,H,DN]
    const __bf16* __restrict__ qr,   // [B,T,H,DR] (rope applied)
    const __bf16* __restrict__ kc,   // [B,T,H,DN]
    const __bf16* __restrict__ kr,   // [B,T,DR]   (rope applied, shared across heads)
    const __bf16* __restrict__ vt,   // [B,H,DV,T]
    float* __restrict__ out) {       // [B,T,H,DV]
  __shared__ __align__(16) __bf16 pshm[4][16][40];  // per-wave 16x32 P tile, 80B stride
  const int lane = threadIdx.x & 31, wv = threadIdx.x >> 5;
  const int l16 = lane & 15, lh = lane >> 4;
  const int q0 = blockIdx.x * 64;
  const int h = blockIdx.y, b = blockIdx.z;
  const float scale = 0.07216878364f;  // 1/sqrt(DN+DR) = 1/sqrt(192)

  // Preload Q fragments for this wave's 16 rows (192 dims -> 6 K-chunks of 32)
  const int qrow = q0 + wv * 16 + l16;
  const __bf16* qcr = qc + (size_t)(b * T_ + qrow) * (H_ * DN_) + h * DN_;
  const __bf16* qrr = qr + (size_t)(b * T_ + qrow) * (H_ * DR_) + h * DR_;
  bf16x16 qf[6];
#pragma unroll
  for (int c = 0; c < 4; ++c) qf[c] = load_frag(qcr + c * 32, lh);
  qf[4] = load_frag(qrr, lh);
  qf[5] = load_frag(qrr + 32, lh);

  f32x8 oacc[8];
#pragma unroll
  for (int i = 0; i < 8; ++i) oacc[i] = (f32x8){};
  float mrow[8], lrow[8];
#pragma unroll
  for (int r = 0; r < 8; ++r) { mrow[r] = -1e30f; lrow[r] = 0.0f; }

  for (int kb = 0; kb < q0 + 64; kb += 32) {
    // ---- S = Q K^T for 2 key subtiles (keys kb..kb+15, kb+16..kb+31) ----
    const __bf16* kc0 = kc + (size_t)(b * T_ + kb + l16) * (H_ * DN_) + h * DN_;
    const __bf16* kc1 = kc0 + (size_t)16 * (H_ * DN_);
    const __bf16* kr0 = kr + (size_t)(b * T_ + kb + l16) * DR_;
    const __bf16* kr1 = kr0 + 16 * DR_;
    f32x8 s0 = {}, s1 = {};
#pragma unroll
    for (int c = 0; c < 4; ++c) {
      s0 = wmma_bf16(qf[c], load_frag(kc0 + c * 32, lh), s0);
      s1 = wmma_bf16(qf[c], load_frag(kc1 + c * 32, lh), s1);
    }
#pragma unroll
    for (int c = 0; c < 2; ++c) {
      s0 = wmma_bf16(qf[4 + c], load_frag(kr0 + c * 32, lh), s0);
      s1 = wmma_bf16(qf[4 + c], load_frag(kr1 + c * 32, lh), s1);
    }

    // ---- causal mask + online softmax (row = r + 8*lh, col = key = l16) ----
    const int kg0 = kb + l16, kg1 = kb + 16 + l16;
#pragma unroll
    for (int r = 0; r < 8; ++r) {
      const int qg = q0 + wv * 16 + 8 * lh + r;
      const float v0 = (kg0 <= qg) ? s0[r] * scale : -1e30f;
      const float v1 = (kg1 <= qg) ? s1[r] * scale : -1e30f;
      float mx = fmaxf(v0, v1);
      mx = fmaxf(mx, __shfl_xor(mx, 1));
      mx = fmaxf(mx, __shfl_xor(mx, 2));
      mx = fmaxf(mx, __shfl_xor(mx, 4));
      mx = fmaxf(mx, __shfl_xor(mx, 8));
      const float mnew = fmaxf(mrow[r], mx);
      const float alpha = __expf(mrow[r] - mnew);
      mrow[r] = mnew;
      const float p0 = __expf(v0 - mnew);
      const float p1 = __expf(v1 - mnew);
      float ps = p0 + p1;
      ps += __shfl_xor(ps, 1);
      ps += __shfl_xor(ps, 2);
      ps += __shfl_xor(ps, 4);
      ps += __shfl_xor(ps, 8);
      lrow[r] = lrow[r] * alpha + ps;
#pragma unroll
      for (int t = 0; t < 8; ++t) oacc[t][r] *= alpha;
      pshm[wv][8 * lh + r][l16] = (__bf16)p0;
      pshm[wv][8 * lh + r][16 + l16] = (__bf16)p1;
    }

    // ---- O += P * V (P re-shaped via per-wave LDS into an A fragment) ----
    bf16x16 pf = load_frag(&pshm[wv][l16][0], lh);
    const __bf16* vbase = vt + (size_t)((b * H_ + h) * DV_ + l16) * T_ + kb;
#pragma unroll
    for (int t = 0; t < 8; ++t) {
      bf16x16 vf = load_frag(vbase + (size_t)(t * 16) * T_, lh);
      oacc[t] = wmma_bf16(pf, vf, oacc[t]);
    }
  }

  // ---- epilogue: normalize and store [B,T,H,DV] ----
#pragma unroll
  for (int r = 0; r < 8; ++r) {
    const float inv = 1.0f / lrow[r];
    const int row = q0 + wv * 16 + 8 * lh + r;
    float* orow = out + (size_t)(b * T_ + row) * (H_ * DV_) + h * DV_;
#pragma unroll
    for (int t = 0; t < 8; ++t) orow[t * 16 + l16] = oacc[t][r] * inv;
  }
}

// ---------------- host-side orchestration ----------------
static inline void* ws_alloc(char* base, size_t& off, size_t bytes) {
  void* p = base + off;
  off = (off + bytes + 255) & ~(size_t)255;
  return p;
}

extern "C" void kernel_launch(void* const* d_in, const int* in_sizes, int n_in,
                              void* d_out, int out_size, void* d_ws, size_t ws_size,
                              hipStream_t stream) {
  (void)in_sizes; (void)n_in; (void)out_size; (void)ws_size;
  const float* x        = (const float*)d_in[0];
  const float* Wq_down  = (const float*)d_in[1];  // [D, QR]
  const float* Wq_up    = (const float*)d_in[2];  // [QR, H*DN]
  const float* Wq_rope  = (const float*)d_in[3];  // [QR, H*DR]
  const float* Wkv_down = (const float*)d_in[4];  // [D, KVR]
  const float* Wk_up    = (const float*)d_in[5];  // [KVR, H*DN]
  const float* Wv_up    = (const float*)d_in[6];  // [KVR, H*DV]
  const float* Wk_rope  = (const float*)d_in[7];  // [D, DR]
  const float* Wo       = (const float*)d_in[8];  // [H*DV, D]
  const int*   positions= (const int*)d_in[9];
  float* out_f = (float*)d_out;

  char* ws = (char*)d_ws;
  size_t off = 0;
  __bf16* xb     = (__bf16*)ws_alloc(ws, off, (size_t)BT_ * D_ * 2);
  __bf16* wqd_t  = (__bf16*)ws_alloc(ws, off, (size_t)QR_ * D_ * 2);
  __bf16* wqu_t  = (__bf16*)ws_alloc(ws, off, (size_t)(H_ * DN_) * QR_ * 2);
  __bf16* wqr_t  = (__bf16*)ws_alloc(ws, off, (size_t)(H_ * DR_) * QR_ * 2);
  __bf16* wkvd_t = (__bf16*)ws_alloc(ws, off, (size_t)KVR_ * D_ * 2);
  __bf16* wku_t  = (__bf16*)ws_alloc(ws, off, (size_t)(H_ * DN_) * KVR_ * 2);
  __bf16* wvu_t  = (__bf16*)ws_alloc(ws, off, (size_t)(H_ * DV_) * KVR_ * 2);
  __bf16* wkr_t  = (__bf16*)ws_alloc(ws, off, (size_t)DR_ * D_ * 2);
  __bf16* wo_t   = (__bf16*)ws_alloc(ws, off, (size_t)D_ * (H_ * DV_) * 2);
  float*  cq_f   = (float*) ws_alloc(ws, off, (size_t)BT_ * QR_ * 4);
  __bf16* cq_b   = (__bf16*)ws_alloc(ws, off, (size_t)BT_ * QR_ * 2);
  float*  ckv_f  = (float*) ws_alloc(ws, off, (size_t)BT_ * KVR_ * 4);
  __bf16* ckv_b  = (__bf16*)ws_alloc(ws, off, (size_t)BT_ * KVR_ * 2);
  float*  qc_f   = (float*) ws_alloc(ws, off, (size_t)BT_ * H_ * DN_ * 4);
  __bf16* qc_b   = (__bf16*)ws_alloc(ws, off, (size_t)BT_ * H_ * DN_ * 2);
  float*  qr_f   = (float*) ws_alloc(ws, off, (size_t)BT_ * H_ * DR_ * 4);
  __bf16* qr_b   = (__bf16*)ws_alloc(ws, off, (size_t)BT_ * H_ * DR_ * 2);
  float*  kc_f   = (float*) ws_alloc(ws, off, (size_t)BT_ * H_ * DN_ * 4);
  __bf16* kc_b   = (__bf16*)ws_alloc(ws, off, (size_t)BT_ * H_ * DN_ * 2);
  float*  vc_f   = (float*) ws_alloc(ws, off, (size_t)BT_ * H_ * DV_ * 4);
  __bf16* vt_b   = (__bf16*)ws_alloc(ws, off, (size_t)B_ * H_ * DV_ * T_ * 2);
  float*  kr_f   = (float*) ws_alloc(ws, off, (size_t)BT_ * DR_ * 4);
  __bf16* kr_b   = (__bf16*)ws_alloc(ws, off, (size_t)BT_ * DR_ * 2);
  float*  ao_f   = (float*) ws_alloc(ws, off, (size_t)BT_ * H_ * DV_ * 4);
  __bf16* ao_b   = (__bf16*)ws_alloc(ws, off, (size_t)BT_ * H_ * DV_ * 2);

  auto cvt = [&](const float* in, __bf16* o, long n) {
    f32_to_bf16_kernel<<<(int)((n + 255) / 256), 256, 0, stream>>>(in, o, n);
  };
  auto wtr = [&](const float* W, __bf16* Wt, int K, int N) {
    wtrans_kernel<<<dim3(N / 32, K / 32), 256, 0, stream>>>(W, Wt, K, N);
  };
  auto gemm = [&](const __bf16* A, const __bf16* Bt, float* C, int M, int N, int K) {
    gemm_bf16_kernel<<<dim3(N / 64, M / 128), 256, 0, stream>>>(A, Bt, C, M, N, K);
  };

  // 0) inputs -> bf16 (weights transposed to [N,K])
  cvt(x, xb, (long)BT_ * D_);
  wtr(Wq_down,  wqd_t,  D_,   QR_);
  wtr(Wq_up,    wqu_t,  QR_,  H_ * DN_);
  wtr(Wq_rope,  wqr_t,  QR_,  H_ * DR_);
  wtr(Wkv_down, wkvd_t, D_,   KVR_);
  wtr(Wk_up,    wku_t,  KVR_, H_ * DN_);
  wtr(Wv_up,    wvu_t,  KVR_, H_ * DV_);
  wtr(Wk_rope,  wkr_t,  D_,   DR_);
  wtr(Wo,       wo_t,   H_ * DV_, D_);

  // 1) query path
  gemm(xb, wqd_t, cq_f, BT_, QR_, D_);
  cvt(cq_f, cq_b, (long)BT_ * QR_);
  gemm(cq_b, wqu_t, qc_f, BT_, H_ * DN_, QR_);
  cvt(qc_f, qc_b, (long)BT_ * H_ * DN_);
  gemm(cq_b, wqr_t, qr_f, BT_, H_ * DR_, QR_);
  rope_q_kernel<<<(B_ * T_ * H_ * (DR_ / 2)) / 256, 256, 0, stream>>>(qr_f, positions, qr_b);

  // 2) kv path
  gemm(xb, wkvd_t, ckv_f, BT_, KVR_, D_);
  cvt(ckv_f, ckv_b, (long)BT_ * KVR_);
  gemm(ckv_b, wku_t, kc_f, BT_, H_ * DN_, KVR_);
  cvt(kc_f, kc_b, (long)BT_ * H_ * DN_);
  gemm(ckv_b, wvu_t, vc_f, BT_, H_ * DV_, KVR_);
  vtrans_kernel<<<(B_ * H_ * DV_ * T_) / 256, 256, 0, stream>>>(vc_f, vt_b);
  gemm(xb, wkr_t, kr_f, BT_, DR_, D_);
  rope_k_kernel<<<(B_ * T_ * (DR_ / 2)) / 256, 256, 0, stream>>>(kr_f, positions, kr_b);

  // 3) causal flash attention
  mla_attn_kernel<<<dim3(T_ / 64, H_, B_), 128, 0, stream>>>(qc_b, qr_b, kc_b, kr_b, vt_b, ao_f);

  // 4) output projection -> d_out (f32 [B,T,D])
  cvt(ao_f, ao_b, (long)BT_ * H_ * DV_);
  gemm(ao_b, wo_t, out_f, BT_, D_, H_ * DV_);
}
